// MyMultiheadAttention_39556648796630
// MI455X (gfx1250) — matrix operand order repaired and checked by reading, not compiled, not run on hardware
//
#include <hip/hip_runtime.h>

// Problem constants (fixed by the reference)
#define BATCH 2
#define SEQ   2048
#define DIM   1024
#define HID   1024
#define HEADS 16
#define HD    64
#define MTOT  (BATCH * SEQ)   // 4096 rows in the projection GEMMs

typedef __attribute__((ext_vector_type(16))) __bf16 v16bf;
typedef __attribute__((ext_vector_type(8)))  __bf16 v8bf;
typedef __attribute__((ext_vector_type(8)))  float  v8f;
typedef __attribute__((ext_vector_type(4)))  int    v4i;

// ---------------------------------------------------------------- helpers ---

__device__ __forceinline__ unsigned short f2bf(float f) {
  union { float f; unsigned int u; } cv; cv.f = f;
  unsigned int u = cv.u;
  u += 0x7fffu + ((u >> 16) & 1u);    // round-to-nearest-even
  return (unsigned short)(u >> 16);
}

// 16-bit A fragment, 16x32 (MxK), row-major source with leading dim `ld`.
// Layout: lanes 0-15 hold row M=lane, elems 0-15 = K 0..15
//         lanes 16-31 hold row M=lane-16, elems = K 16..31 (matches load_frag_b)
__device__ __forceinline__ v16bf load_frag_a(const unsigned short* base, int ld,
                                             int k0, int lane) {
  const int half = lane >> 4;
  const int row  = lane & 15;
  const unsigned short* p = base + (size_t)row * ld + k0 + half * 16;
  v8bf lo = *(const v8bf*)(p);
  v8bf hi = *(const v8bf*)(p + 8);
  return __builtin_shufflevector(lo, hi, 0,1,2,3,4,5,6,7,8,9,10,11,12,13,14,15);
}

// 16-bit B fragment, 32x16 (KxN) where B[k][n] = W[n][k]; W row-major [N,K].
// Layout used here (contiguous k): lanes 0-15 hold col N=lane, K = k0..k0+15 ; lanes 16-31: K += 16.
// That is a contiguous 32-byte slice of W's row n.
__device__ __forceinline__ v16bf load_frag_b(const unsigned short* wrow, int ld,
                                             int k0, int lane) {
  const int half = lane >> 4;
  const int n    = lane & 15;
  return *(const v16bf*)(wrow + (size_t)n * ld + k0 + half * 16);
}

__device__ __forceinline__ v8f wmma_bf16(v16bf a, v16bf b, v8f c) {
  return __builtin_amdgcn_wmma_f32_16x16x32_bf16(false, a, false, b,
                                                 (short)0, c, false, false);
}

// --------------------------------------------------- async copy to LDS ------
// 16-byte async global->LDS copy (ASYNCcnt-tracked, completes in order).
// Builtin signature (from compiler diagnostic): param 1 is a non-const
// int4 pointer in the global (__device__) address space; param 2 is the LDS
// destination; then (imm offset, imm cpol).

typedef __attribute__((address_space(1))) v4i* gptr_v4i;
typedef __attribute__((address_space(3))) v4i* lptr_v4i;

__device__ __forceinline__ void async_cp16(unsigned short* lp,
                                           const unsigned short* gp) {
#if __has_builtin(__builtin_amdgcn_global_load_async_to_lds_b128)
  __builtin_amdgcn_global_load_async_to_lds_b128((gptr_v4i)gp, (lptr_v4i)lp, 0, 0);
#else
  __attribute__((address_space(3))) unsigned short* l3 =
      (__attribute__((address_space(3))) unsigned short*)lp;
  asm volatile("global_load_async_to_lds_b128 %0, %1, off"
               :
               : "v"(l3), "v"(gp)
               : "memory");
#endif
}

__device__ __forceinline__ void wait_async0() {
#if __has_builtin(__builtin_amdgcn_s_wait_asynccnt)
  __builtin_amdgcn_s_wait_asynccnt(0);
#else
  asm volatile("s_wait_asynccnt 0x0" ::: "memory");
#endif
}

// ------------------------------------------------------------ cvt kernel ----

__global__ void __launch_bounds__(256)
cvt_f32_bf16_kernel(const float* __restrict__ src, unsigned short* __restrict__ dst,
                    int n) {
  int idx = (blockIdx.x * blockDim.x + threadIdx.x) * 4;
  if (idx < n) {
    float4 v = *(const float4*)(src + idx);
    ushort4 o;
    o.x = f2bf(v.x); o.y = f2bf(v.y); o.z = f2bf(v.z); o.w = f2bf(v.w);
    *(ushort4*)(dst + idx) = o;
  }
}

// ------------------------------------------------------------ GEMM kernel ---
// C[M,N] = A[M,K] * W[N,K]^T + bias.  M=4096, N=1024, K=1024.
// Block: 256 threads = 8 waves, block tile 64M x 256N, wave tile 32M x 64N.
// mode 0: out bf16 [B,H,S,hd]   (Q, K projections)
// mode 1: out bf16 [B,H,hd,S]   (V projection, transposed for PV B-fragments)
// mode 2: out fp32 [M,N]        (final output projection -> d_out)

__global__ void __launch_bounds__(256)
gemm_bf16_kernel(const unsigned short* __restrict__ A,
                 const unsigned short* __restrict__ W,
                 const float* __restrict__ bias,
                 void* __restrict__ out, int mode) {
  const int lane = threadIdx.x & 31;
  const int wave = threadIdx.x >> 5;
  const int wm = wave & 1;          // 2 waves along M
  const int wn = wave >> 1;         // 4 waves along N
  const int row0 = blockIdx.x * 64 + wm * 32;
  const int col0 = blockIdx.y * 256 + wn * 64;
  const int K = DIM;

  v8f c[2][4] = {};

  const unsigned short* Arow0 = A + (size_t)row0 * K;
  const unsigned short* Arow1 = A + (size_t)(row0 + 16) * K;

  for (int k0 = 0; k0 < K; k0 += 32) {
    if (k0 + 32 < K) {    // warm next K-slice of A (global_prefetch_b8)
      __builtin_prefetch(Arow0 + (size_t)(lane & 15) * K + k0 + 32, 0, 1);
    }
    v16bf a0 = load_frag_a(Arow0, K, k0, lane);
    v16bf a1 = load_frag_a(Arow1, K, k0, lane);
#pragma unroll
    for (int j = 0; j < 4; ++j) {
      v16bf b = load_frag_b(W + (size_t)(col0 + 16 * j) * K, K, k0, lane);
      c[0][j] = wmma_bf16(a0, b, c[0][j]);
      c[1][j] = wmma_bf16(a1, b, c[1][j]);
    }
  }

  // C fragment layout: VGPR r -> M = r + 8*(lane/16), N = lane%16
  const int half = lane >> 4;
  const int ln   = lane & 15;
#pragma unroll
  for (int i = 0; i < 2; ++i) {
#pragma unroll
    for (int j = 0; j < 4; ++j) {
      const int gc = col0 + 16 * j + ln;
      const float bv = bias[gc];
#pragma unroll
      for (int r = 0; r < 8; ++r) {
        const int gr = row0 + 16 * i + r + 8 * half;
        const float v = c[i][j][r] + bv;
        if (mode == 2) {
          ((float*)out)[(size_t)gr * HID + gc] = v;
        } else {
          const int bb = gr >> 11;            // / SEQ
          const int s  = gr & (SEQ - 1);
          const int h  = gc >> 6;             // / HD
          const int d  = gc & (HD - 1);
          size_t addr;
          if (mode == 0) addr = (((size_t)bb * HEADS + h) * SEQ + s) * HD + d;
          else           addr = (((size_t)bb * HEADS + h) * HD + d) * (size_t)SEQ + s;
          ((unsigned short*)out)[addr] = f2bf(v);
        }
      }
    }
  }
}

// ------------------------------------------------------- attention kernel ---
// Grid: B*H*(S/64) blocks of 128 threads (4 waves). Each wave owns 16 queries;
// the block streams 32 keys/iter. K/V tiles are staged cooperatively into LDS
// with double-buffered GLOBAL_LOAD_ASYNC_TO_LDS_B128 (all 4 waves share them),
// overlapping the async engine's fetch of tile i+1 with compute on tile i.
// scale = 1/sqrt(HID) = 1/32 (reference quirk: scales by hidden_dim).

__device__ __forceinline__ void stage_tiles(const unsigned short* Kb,
                                            const unsigned short* Vb,
                                            int kb, unsigned short* kt,
                                            unsigned short* vt, int t) {
  // K tile: rows kb..kb+31 of [S,64] are one contiguous 4KB chunk.
  async_cp16(kt + t * 8,        Kb + (size_t)kb * HD + t * 8);
  async_cp16(kt + 1024 + t * 8, Kb + (size_t)kb * HD + 1024 + t * 8);
  // V tile: 64 rows of VT [hd,S], 64B per row (cols kb..kb+31).
#pragma unroll
  for (int q = 0; q < 2; ++q) {
    const int idx = t + q * 128;
    const int row = idx >> 2;
    const int part = idx & 3;
    async_cp16(vt + row * 32 + part * 8, Vb + (size_t)row * SEQ + kb + part * 8);
  }
}

__global__ void __launch_bounds__(128)
attn_kernel(const unsigned short* __restrict__ Q,
            const unsigned short* __restrict__ Kmat,
            const unsigned short* __restrict__ VT,
            unsigned short* __restrict__ ctx) {
  __shared__ __align__(16) unsigned short KtLds[2][32][64];  // keys x d
  __shared__ __align__(16) unsigned short VtLds[2][64][32];  // d x keys
  __shared__ __align__(16) unsigned short Plds[4][16][32];   // per-wave P tile

  const int t    = threadIdx.x;
  const int lane = t & 31;
  const int wave = t >> 5;
  const int bh   = blockIdx.x >> 5;     // S/64 = 32 query blocks per (b,h)
  const int qblk = blockIdx.x & 31;
  const int b = bh >> 4;
  const int h = bh & 15;

  const unsigned short* Qb = Q    + (size_t)bh * SEQ * HD;
  const unsigned short* Kb = Kmat + (size_t)bh * SEQ * HD;
  const unsigned short* Vb = VT   + (size_t)bh * HD * SEQ;

  const int q0 = qblk * 64 + wave * 16;
  const v16bf qa0 = load_frag_a(Qb + (size_t)q0 * HD, HD, 0,  lane);
  const v16bf qa1 = load_frag_a(Qb + (size_t)q0 * HD, HD, 32, lane);

  const int half = lane >> 4;
  const int ln   = lane & 15;
  const float SCALE = 0.03125f;  // 1/sqrt(1024)

  float mrun[8], lrun[8];
  v8f acc[4] = {};
#pragma unroll
  for (int r = 0; r < 8; ++r) { mrun[r] = -3.0e38f; lrun[r] = 0.0f; }

  stage_tiles(Kb, Vb, 0, &KtLds[0][0][0], &VtLds[0][0][0], t);

  for (int kb = 0, it = 0; kb < SEQ; kb += 32, ++it) {
    const int buf = it & 1;
    wait_async0();        // this wave's tile `it` is resident
    __syncthreads();      // resident for all waves; old buffer free to overwrite
    if (kb + 32 < SEQ) {  // async engine fetches tile it+1 while we compute
      stage_tiles(Kb, Vb, kb + 32, &KtLds[buf ^ 1][0][0], &VtLds[buf ^ 1][0][0], t);
    }

    const unsigned short* kt = &KtLds[buf][0][0];

    // scores: 16 queries x 32 keys, inner dim hd=64 (two K=32 WMMA steps)
    v8f s0 = {}, s1 = {};
    s0 = wmma_bf16(qa0, load_frag_b(kt,            HD, 0,  lane), s0);
    s0 = wmma_bf16(qa1, load_frag_b(kt,            HD, 32, lane), s0);
    s1 = wmma_bf16(qa0, load_frag_b(kt + 16 * HD,  HD, 0,  lane), s1);
    s1 = wmma_bf16(qa1, load_frag_b(kt + 16 * HD,  HD, 32, lane), s1);

    float corr[8];
#pragma unroll
    for (int r = 0; r < 8; ++r) {
      const float a  = s0[r] * SCALE;
      const float bt = s1[r] * SCALE;
      // row max over 32 keys: local pair max + xor-reduce across the 16-lane
      // half that shares this row (masks < 16 stay within a half)
      float mx = fmaxf(a, bt);
      mx = fmaxf(mx, __shfl_xor(mx, 1, 32));
      mx = fmaxf(mx, __shfl_xor(mx, 2, 32));
      mx = fmaxf(mx, __shfl_xor(mx, 4, 32));
      mx = fmaxf(mx, __shfl_xor(mx, 8, 32));
      const float mnew = fmaxf(mrun[r], mx);
      const float cr = __expf(mrun[r] - mnew);
      const float p0 = __expf(a  - mnew);
      const float p1 = __expf(bt - mnew);
      float rs = p0 + p1;
      rs += __shfl_xor(rs, 1, 32);
      rs += __shfl_xor(rs, 2, 32);
      rs += __shfl_xor(rs, 4, 32);
      rs += __shfl_xor(rs, 8, 32);
      lrun[r] = lrun[r] * cr + rs;
      mrun[r] = mnew;
      corr[r] = cr;
      // stash P (C-layout) into per-wave LDS so it can be re-read in A-layout
      Plds[wave][r + 8 * half][ln]      = f2bf(p0);
      Plds[wave][r + 8 * half][16 + ln] = f2bf(p1);
    }
#pragma unroll
    for (int j = 0; j < 4; ++j)
#pragma unroll
      for (int r = 0; r < 8; ++r) acc[j][r] *= corr[r];

    __builtin_amdgcn_wave_barrier();   // order LDS stores before A-frag reload
    const v16bf pa = load_frag_a(&Plds[wave][0][0], 32, 0, lane);
    __builtin_amdgcn_wave_barrier();   // keep next iter's stores after this load

    // out += P(16x32) x V(32x64): B-frags = contiguous rows of the LDS V tile
#pragma unroll
    for (int j = 0; j < 4; ++j) {
      const v16bf vb = *(const v16bf*)(&VtLds[buf][16 * j + ln][half * 16]);
      acc[j] = wmma_bf16(pa, vb, acc[j]);
    }
  }

  // epilogue: ctx[b*S + s][h*64 + d] = acc / l
#pragma unroll
  for (int j = 0; j < 4; ++j) {
#pragma unroll
    for (int r = 0; r < 8; ++r) {
      const int s = q0 + r + 8 * half;
      const float o = acc[j][r] / lrun[r];
      ctx[((size_t)b * SEQ + s) * HID + h * HD + 16 * j + ln] = f2bf(o);
    }
  }
}

// --------------------------------------------------------------- launcher ---

extern "C" void kernel_launch(void* const* d_in, const int* in_sizes, int n_in,
                              void* d_out, int out_size, void* d_ws, size_t ws_size,
                              hipStream_t stream) {
  const float* x  = (const float*)d_in[0];
  const float* Wq = (const float*)d_in[1];
  const float* bq = (const float*)d_in[2];
  const float* Wk = (const float*)d_in[3];
  const float* bk = (const float*)d_in[4];
  const float* Wv = (const float*)d_in[5];
  const float* bv = (const float*)d_in[6];
  const float* Wo = (const float*)d_in[7];
  const float* bo = (const float*)d_in[8];
  float* out = (float*)d_out;

  // workspace carve-up (48 MB total)
  char* p = (char*)d_ws;
  unsigned short* xb  = (unsigned short*)p; p += (size_t)MTOT * DIM * 2;
  unsigned short* wqb = (unsigned short*)p; p += (size_t)HID * DIM * 2;
  unsigned short* wkb = (unsigned short*)p; p += (size_t)HID * DIM * 2;
  unsigned short* wvb = (unsigned short*)p; p += (size_t)HID * DIM * 2;
  unsigned short* wob = (unsigned short*)p; p += (size_t)DIM * HID * 2;
  unsigned short* Qb  = (unsigned short*)p; p += (size_t)MTOT * HID * 2;
  unsigned short* Kb  = (unsigned short*)p; p += (size_t)MTOT * HID * 2;
  unsigned short* VTb = (unsigned short*)p; p += (size_t)MTOT * HID * 2;
  unsigned short* ctx = (unsigned short*)p; p += (size_t)MTOT * HID * 2;

  // 1) fp32 -> bf16 conversions
  {
    const int nx = MTOT * DIM;
    cvt_f32_bf16_kernel<<<nx / 4 / 256, 256, 0, stream>>>(x, xb, nx);
    const int nw = HID * DIM;
    cvt_f32_bf16_kernel<<<nw / 4 / 256, 256, 0, stream>>>(Wq, wqb, nw);
    cvt_f32_bf16_kernel<<<nw / 4 / 256, 256, 0, stream>>>(Wk, wkb, nw);
    cvt_f32_bf16_kernel<<<nw / 4 / 256, 256, 0, stream>>>(Wv, wvb, nw);
    cvt_f32_bf16_kernel<<<nw / 4 / 256, 256, 0, stream>>>(Wo, wob, nw);
  }

  // 2) QKV projections (V transposed per-head for PV B-fragment loads)
  dim3 ggrid(MTOT / 64, HID / 256);
  gemm_bf16_kernel<<<ggrid, 256, 0, stream>>>(xb, wqb, bq, Qb, 0);
  gemm_bf16_kernel<<<ggrid, 256, 0, stream>>>(xb, wkb, bk, Kb, 0);
  gemm_bf16_kernel<<<ggrid, 256, 0, stream>>>(xb, wvb, bv, VTb, 1);

  // 3) flash attention (async double-buffered K/V staging)
  attn_kernel<<<BATCH * HEADS * (SEQ / 64), 128, 0, stream>>>(Qb, Kb, VTb, ctx);

  // 4) output projection -> fp32 d_out
  gemm_bf16_kernel<<<ggrid, 256, 0, stream>>>(ctx, wob, bo, out, 2);
}
